// PairEmbedding_9285719294279
// MI455X (gfx1250) — matrix operand order repaired and checked by reading, not compiled, not run on hardware
//
#include <hip/hip_runtime.h>
#include <hip/hip_bf16.h>

typedef _Float16 v16h __attribute__((ext_vector_type(16)));
typedef _Float16 v8h  __attribute__((ext_vector_type(8)));
typedef float    v8f  __attribute__((ext_vector_type(8)));

#define WAVES 4

// Fast SiLU: x * v_rcp_f32(1 + exp(-x)). Avoids the IEEE div expansion
// (v_div_scale + NR chain) the compiler emits for plain '/'.
__device__ __forceinline__ float silu_f(float x) {
    return x * __builtin_amdgcn_rcpf(1.0f + __expf(-x));
}

// A-fragment for V_WMMA_F32_16X16X32_F16 from row-major LDS [16][128] f16.
// Lane l, halfs j:  j<8 -> k = k0 + (l/16)*8 + j ;  j>=8 -> +16.
// => two contiguous 16B chunks at (k0 + (l/16)*8) and +16.
__device__ __forceinline__ v16h load_a_frag(const _Float16* act, int k0, int lane) {
    const int m    = lane & 15;
    const int base = k0 + ((lane >> 4) << 3);
    const _Float16* p = act + m * 128 + base;
    union { v16h v; v8h h[2]; } u;
    u.h[0] = *(const v8h*)(p);
    u.h[1] = *(const v8h*)(p + 16);
    return u.v;
}

// One chunk of NT 16x16 output tiles: C[nt] = A(16xK) * B(K x ...), K = KT*32.
// B-fragments are explicitly double-buffered: frags for kt+1 are loaded while
// the WMMAs for kt execute, so no WMMA waits on its own load. Accumulators
// start at 0 (inline SRC2=0); bias is folded into the epilogue.
// NTtot = total N tiles of the layer (fragment stride); ntBase = chunk origin.
template<int KT, int NT>
__device__ __forceinline__ void gemm_chunk(const _Float16* act, const _Float16* wp,
                                           int lane, int ntBase, int NTtot, v8f c[NT]) {
#pragma unroll
    for (int nt = 0; nt < NT; ++nt)
#pragma unroll
        for (int i = 0; i < 8; ++i) c[nt][i] = 0.0f;

    const _Float16* base = wp + (((ntBase * 32) + lane) << 4);
    v16h b[NT];
#pragma unroll
    for (int nt = 0; nt < NT; ++nt)
        b[nt] = *(const v16h*)(base + nt * 512);

#pragma unroll
    for (int kt = 0; kt < KT; ++kt) {
        const v16h a = load_a_frag(act, kt * 32, lane);
        v16h bn[NT];
        if (kt + 1 < KT) {
            const _Float16* nb = base + (size_t)(kt + 1) * NTtot * 512;
#pragma unroll
            for (int nt = 0; nt < NT; ++nt)
                bn[nt] = *(const v16h*)(nb + nt * 512);
        }
#pragma unroll
        for (int nt = 0; nt < NT; ++nt)
            c[nt] = __builtin_amdgcn_wmma_f32_16x16x32_f16(false, a, false, b[nt],
                                                           (short)0, c[nt],
                                                           false, false);
        if (kt + 1 < KT) {
#pragma unroll
            for (int nt = 0; nt < NT; ++nt) b[nt] = bn[nt];
        }
    }
}

// SiLU + bias epilogue for a 4-tile chunk -> f16 LDS activations.
__device__ __forceinline__ void epi_silu(const v8f c[4], const float* __restrict__ bias,
                                         _Float16 (*dst)[128], int ntBase, int lane) {
    const int n16 = lane & 15;
    const int mh  = (lane >> 4) * 8;
#pragma unroll
    for (int q = 0; q < 4; ++q) {
        const int nt   = ntBase + q;
        const float bv = bias[nt * 16 + n16];
        const int   n  = nt * 16 + n16;
#pragma unroll
        for (int v = 0; v < 8; ++v)
            dst[mh + v][n] = (_Float16)silu_f(c[q][v] + bv);
    }
}

// Pack f32 weights (Kreal x N, row-major) into WMMA B-fragment order, f16,
// padded to Kp rows with zeros. Fragment (kt,nt): lane l, half j holds
// w[kt*32 + (l/16)*8 + (j&7) + (j/8)*16][nt*16 + l%16].
__global__ void pack_weights_kernel(const float* __restrict__ w, _Float16* __restrict__ wp,
                                    int Kreal, int Kp, int N) {
    const int kTiles = Kp >> 5, nTiles = N >> 4;
    const int total  = kTiles * nTiles * 512;
    for (int idx = blockIdx.x * blockDim.x + threadIdx.x; idx < total;
         idx += gridDim.x * blockDim.x) {
        const int j    = idx & 15;
        const int lane = (idx >> 4) & 31;
        const int frag = idx >> 9;
        const int nt   = frag % nTiles;
        const int kt   = frag / nTiles;
        const int k    = kt * 32 + ((lane >> 4) << 3) + (j & 7) + ((j >> 3) << 4);
        const int n    = nt * 16 + (lane & 15);
        const float v  = (k < Kreal) ? w[k * N + n] : 0.0f;
        wp[idx] = (_Float16)v;
    }
}

__global__ __launch_bounds__(WAVES * 32) void pair_embed_kernel(
    const int* __restrict__ Z, const float* __restrict__ bond_dist,
    const float* __restrict__ bond_diff, const int* __restrict__ edge_index,
    const float* __restrict__ emb,
    const float* __restrict__ b0, const float* __restrict__ b1,
    const float* __restrict__ b2, const float* __restrict__ be,
    const _Float16* __restrict__ w0p, const _Float16* __restrict__ w1p,
    const _Float16* __restrict__ w2p, const _Float16* __restrict__ wep,
    float* __restrict__ out_scalar, float* __restrict__ out_tensor,
    float* __restrict__ out_ang, float* __restrict__ out_cut, int nTiles) {

    __shared__ __align__(16) _Float16 actA[WAVES][16][128];
    __shared__ __align__(16) _Float16 actB[WAVES][16][128];
    __shared__ float shv[WAVES][16][9];

    const int lane = threadIdx.x & 31;
    const int wave = threadIdx.x >> 5;
    const int tile = blockIdx.x * WAVES + wave;
    if (tile >= nTiles) return;
    const int e0 = tile << 4;

    // Warm WGP$/L2 with the packed weight region (90112 B, contiguous in ws).
    for (int off = (int)threadIdx.x * 256; off < 90112; off += WAVES * 32 * 256)
        __builtin_prefetch((const char*)w0p + off, 0, 3);

    _Float16 (*A)[128] = actA[wave];
    _Float16 (*B)[128] = actB[wave];
    float    (*SH)[9]  = shv[wave];

    // ---------------- per-edge prep: features 0..63 into A, SH into LDS ----
    {
        const int m = lane & 15;
        const int e = e0 + m;
        const float r = bond_dist[e];
        if (lane < 16) {
            // poly cutoff, p=6: 1 - 28u^6 + 48u^7 - 21u^8
            const float u  = r * 0.2f;
            const float u2 = u * u, u3 = u2 * u;
            const float u6 = u3 * u3, u7 = u6 * u, u8 = u7 * u;
            float fc = 1.0f - 28.0f * u6 + 48.0f * u7 - 21.0f * u8;
            fc = (u < 1.0f) ? fc : 0.0f;
            out_cut[e] = fc;
            // bessel rbf * cutoff -> features 0..7
            const float pref  = 0.6324555320336759f;          // sqrt(2/5)
            const float inv_r = __builtin_amdgcn_rcpf(r);
            const float w     = 0.6283185307179586f * r;      // pi*r/5
            const float s     = pref * inv_r * fc;
#pragma unroll
            for (int n = 0; n < 8; ++n)
                A[m][n] = (_Float16)(s * __sinf((float)(n + 1) * w));
            // src type embedding -> features 8..23
            const int zs = Z[edge_index[2 * e + 0]];
            const float* er = emb + zs * 16;
#pragma unroll
            for (int i = 0; i < 16; ++i) A[m][8 + i] = (_Float16)er[i];
        } else {
            // dst type embedding -> features 24..39, zero-pad 40..63
            const int zd = Z[edge_index[2 * e + 1]];
            const float* er = emb + zd * 16;
#pragma unroll
            for (int i = 0; i < 16; ++i) A[m][24 + i] = (_Float16)er[i];
#pragma unroll
            for (int i = 40; i < 64; ++i) A[m][i] = (_Float16)0.0f;
            // spherical harmonics of unit direction
            const float invr = __builtin_amdgcn_rcpf(r + 1e-8f);
            const float x = bond_diff[3 * e + 0] * invr;
            const float y = bond_diff[3 * e + 1] * invr;
            const float z = bond_diff[3 * e + 2] * invr;
            const float s3 = 1.7320508075688772f;
            const float s15 = 3.872983346207417f;
            const float s5 = 2.23606797749979f;
            float sh[9];
            sh[0] = 1.0f;        sh[1] = s3 * x;   sh[2] = s3 * y;  sh[3] = s3 * z;
            sh[4] = s15 * x * y; sh[5] = s15 * y * z;
            sh[6] = 0.5f * s5 * (3.0f * z * z - 1.0f);
            sh[7] = s15 * x * z; sh[8] = 0.5f * s15 * (x * x - y * y);
            float* ao = out_ang + (size_t)e * 9;
#pragma unroll
            for (int i = 0; i < 9; ++i) { SH[m][i] = sh[i]; ao[i] = sh[i]; }
        }
    }
    // Same-wave DS ordering makes prep stores visible to the GEMM loads.

    const int n16 = lane & 15;
    const int mh  = (lane >> 4) * 8;

    // ---- L1: [16x64] x [64x128] + b0, SiLU -> B  (16 WMMA) ----
    {
        v8f c[4];
        gemm_chunk<2, 4>(&A[0][0], w0p, lane, 0, 8, c);
        epi_silu(c, b0, B, 0, lane);
        gemm_chunk<2, 4>(&A[0][0], w0p, lane, 4, 8, c);
        epi_silu(c, b0, B, 4, lane);
    }
    // ---- L2: [16x128] x [128x128] + b1, SiLU -> A  (32 WMMA) ----
    {
        v8f c[4];
        gemm_chunk<4, 4>(&B[0][0], w1p, lane, 0, 8, c);
        epi_silu(c, b1, A, 0, lane);
        gemm_chunk<4, 4>(&B[0][0], w1p, lane, 4, 8, c);
        epi_silu(c, b1, A, 4, lane);
    }
    // ---- L3: [16x128] x [128x128] + b2, SiLU -> B + scalar out  (32 WMMA) --
#pragma unroll
    for (int half = 0; half < 2; ++half) {
        v8f c[4];
        gemm_chunk<4, 4>(&A[0][0], w2p, lane, half * 4, 8, c);
#pragma unroll
        for (int q = 0; q < 4; ++q) {
            const int nt   = half * 4 + q;
            const float bv = b2[nt * 16 + n16];
            const int   n  = nt * 16 + n16;
#pragma unroll
            for (int v = 0; v < 8; ++v) {
                const float x = silu_f(c[q][v] + bv);
                B[mh + v][n] = (_Float16)x;
                out_scalar[(size_t)(e0 + mh + v) * 128 + n] = x;
            }
        }
    }
    // ---- L4: env = [16x128] x [128x32] + be; tensor = SH (x) env (8 WMMA) --
    {
        v8f c[2];
        gemm_chunk<4, 2>(&B[0][0], wep, lane, 0, 2, c);
#pragma unroll
        for (int nt = 0; nt < 2; ++nt) {
            const float bv = be[nt * 16 + n16];
            const int   n  = nt * 16 + n16;
#pragma unroll
            for (int v = 0; v < 8; ++v) {
                const int m = mh + v;
                const float env = c[nt][v] + bv;
                float* to = out_tensor + (size_t)(e0 + m) * 288 + n;
#pragma unroll
                for (int i = 0; i < 9; ++i) to[i * 32] = SH[m][i] * env;
            }
        }
    }
}

extern "C" void kernel_launch(void* const* d_in, const int* in_sizes, int n_in,
                              void* d_out, int out_size, void* d_ws, size_t ws_size,
                              hipStream_t stream) {
    const int*   Z         = (const int*)d_in[0];
    const float* bond_dist = (const float*)d_in[1];
    const float* bond_diff = (const float*)d_in[2];
    const int*   edge_idx  = (const int*)d_in[3];
    const float* emb       = (const float*)d_in[4];
    const float* w0 = (const float*)d_in[5];  const float* b0 = (const float*)d_in[6];
    const float* w1 = (const float*)d_in[7];  const float* b1 = (const float*)d_in[8];
    const float* w2 = (const float*)d_in[9];  const float* b2 = (const float*)d_in[10];
    const float* we = (const float*)d_in[11]; const float* be = (const float*)d_in[12];

    const int nE = in_sizes[1];                 // bond_dist element count
    float* out        = (float*)d_out;
    float* out_scalar = out;
    float* out_tensor = out_scalar + (size_t)nE * 128;
    float* out_ang    = out_tensor + (size_t)nE * 288;
    float* out_cut    = out_ang    + (size_t)nE * 9;

    // Workspace: packed f16 weight fragments (90112 bytes total).
    _Float16* ws  = (_Float16*)d_ws;
    _Float16* w0p = ws;                 // 2kt * 8nt * 512 = 8192 halfs
    _Float16* w1p = ws + 8192;          // 4kt * 8nt * 512 = 16384
    _Float16* w2p = ws + 24576;         // 16384
    _Float16* wep = ws + 40960;         // 4kt * 2nt * 512 = 4096

    pack_weights_kernel<<<32, 256, 0, stream>>>(w0, w0p, 40, 64, 128);
    pack_weights_kernel<<<64, 256, 0, stream>>>(w1, w1p, 128, 128, 128);
    pack_weights_kernel<<<64, 256, 0, stream>>>(w2, w2p, 128, 128, 128);
    pack_weights_kernel<<<16, 256, 0, stream>>>(we, wep, 128, 128, 32);

    const int nTiles = nE >> 4;                 // 640000/16 = 40000
    const int blocks = (nTiles + WAVES - 1) / WAVES;
    pair_embed_kernel<<<blocks, WAVES * 32, 0, stream>>>(
        Z, bond_dist, bond_diff, edge_idx, emb, b0, b1, b2, be,
        w0p, w1p, w2p, wep, out_scalar, out_tensor, out_ang, out_cut, nTiles);
}
